// AttentiveFPModel_3023656976500
// MI455X (gfx1250) — compile-verified, matrix operand order checked by reading, not run on hardware
//
#include <hip/hip_runtime.h>
#include <hip/hip_bf16.h>
#include <math.h>
#include <stdint.h>

// ---------------- constants (from reference) ----------------
#define Hdim   256
#define Gnum   8
#define OUTdim 64

typedef __attribute__((ext_vector_type(16))) __bf16 v16bf;
typedef __attribute__((ext_vector_type(8)))  __bf16 v8bf;
typedef __attribute__((ext_vector_type(8)))  float  v8f;

// ---------------- device helpers ----------------
__device__ __forceinline__ float lrelu_f(float v) { return v >= 0.f ? v : 0.01f * v; }
__device__ __forceinline__ float sigm_f(float v)  { return 1.f / (1.f + __expf(-v)); }

// order-preserving float<->uint encoding for atomicMax-based segment max
__device__ __forceinline__ unsigned fenc(float f) {
  unsigned u = __float_as_uint(f);
  return (u & 0x80000000u) ? ~u : (u | 0x80000000u);
}
__device__ __forceinline__ float fdec(unsigned u) {
  return __uint_as_float((u & 0x80000000u) ? (u & 0x7fffffffu) : ~u);
}
#define ENC_NEG_INF 0x007FFFFFu  // fenc(-inf)

// ---------------- WMMA fragment helpers ----------------
// A fragment (16x32 bf16, wave32): lane L holds row M = L%16; reg r holds K pair
//   kb = (r&3)*2 + (L>=16)*8 + (r>=4)*16   (CDNA5 ISA 7.12.2)
// B fragment (32x16): mirror of A indexed by N (B = W^T, lane reads W row N).
// bf16 row-major memory: fragment = two 16B chunks -> two b128 loads, zero VALU.
struct RawH { v8bf lo, hi; };
__device__ __forceinline__ RawH load_raw_h(const __bf16* __restrict__ row, int kk, int sel) {
  RawH r;
  r.lo = *(const v8bf*)(row + kk + (sel << 3));
  r.hi = *(const v8bf*)(row + kk + 16 + (sel << 3));
  return r;
}
__device__ __forceinline__ v16bf cat_frag(const RawH& r) {
  v16bf f;
#pragma unroll
  for (int i = 0; i < 8; ++i) { f[i] = r.lo[i]; f[8 + i] = r.hi[i]; }
  return f;
}
// fp32-A path (stage 0 only: input x is fp32)
struct RawF { float4 a0, a1, b0, b1; };
__device__ __forceinline__ RawF load_raw_f(const float* __restrict__ row, int kk, int sel) {
  RawF r;
  const float4* p = (const float4*)(row + kk + (sel << 3));
  r.a0 = p[0]; r.a1 = p[1];
  const float4* q = (const float4*)(row + kk + 16 + (sel << 3));
  r.b0 = q[0]; r.b1 = q[1];
  return r;
}
__device__ __forceinline__ v16bf cvt_frag(const RawF& r) {
  v16bf f;
  f[0]  = (__bf16)r.a0.x; f[1]  = (__bf16)r.a0.y;
  f[2]  = (__bf16)r.a0.z; f[3]  = (__bf16)r.a0.w;
  f[4]  = (__bf16)r.a1.x; f[5]  = (__bf16)r.a1.y;
  f[6]  = (__bf16)r.a1.z; f[7]  = (__bf16)r.a1.w;
  f[8]  = (__bf16)r.b0.x; f[9]  = (__bf16)r.b0.y;
  f[10] = (__bf16)r.b0.z; f[11] = (__bf16)r.b0.w;
  f[12] = (__bf16)r.b1.x; f[13] = (__bf16)r.b1.y;
  f[14] = (__bf16)r.b1.z; f[15] = (__bf16)r.b1.w;
  return f;
}

// ---------------- WMMA GEMM (bf16 A, bf16 W): C = act(A @ W^T + bias) ----------
// block = 256 threads = 8 waves; wave w owns C rows [m0,m0+16) x cols [n0,n0+64).
// Software-pipelined: next k-step's raw loads issued before this step's wmmas.
__global__ __launch_bounds__(256)
void gemm_wmma_hh(const __bf16* __restrict__ A, int lda,
                  const __bf16* __restrict__ Wt, int ldw,
                  const float* __restrict__ bias,
                  float* __restrict__ C, __bf16* __restrict__ Cb,
                  int M, int Nout, int K, int act) {
  const int wave = threadIdx.x >> 5;
  const int lane = threadIdx.x & 31;
  const int m0 = (blockIdx.x * 8 + wave) * 16;
  const int n0 = blockIdx.y * 64;
  const int sel = lane >> 4;
  const int ra = min(m0 + (lane & 15), M - 1);   // clamped; rows >= M masked at store
  const __bf16* arow = A + (size_t)ra * lda;
  const __bf16* brow0 = Wt + (size_t)(n0 + (lane & 15)) * ldw;   // Nout % 64 == 0
  const __bf16* brow1 = brow0 + (size_t)16 * ldw;
  const __bf16* brow2 = brow0 + (size_t)32 * ldw;
  const __bf16* brow3 = brow0 + (size_t)48 * ldw;

  v8f acc0 = {}, acc1 = {}, acc2 = {}, acc3 = {};
  RawH rA  = load_raw_h(arow,  0, sel);
  RawH rB0 = load_raw_h(brow0, 0, sel);
  RawH rB1 = load_raw_h(brow1, 0, sel);
  RawH rB2 = load_raw_h(brow2, 0, sel);
  RawH rB3 = load_raw_h(brow3, 0, sel);
  for (int kk = 0; kk < K; kk += 32) {
    v16bf af = cat_frag(rA);
    v16bf f0 = cat_frag(rB0);
    v16bf f1 = cat_frag(rB1);
    v16bf f2 = cat_frag(rB2);
    v16bf f3 = cat_frag(rB3);
    const int kn = (kk + 32 < K) ? kk + 32 : kk;
    rA  = load_raw_h(arow,  kn, sel);
    rB0 = load_raw_h(brow0, kn, sel);
    rB1 = load_raw_h(brow1, kn, sel);
    rB2 = load_raw_h(brow2, kn, sel);
    rB3 = load_raw_h(brow3, kn, sel);
    acc0 = __builtin_amdgcn_wmma_f32_16x16x32_bf16(false, af, false, f0, (short)0, acc0, false, false);
    acc1 = __builtin_amdgcn_wmma_f32_16x16x32_bf16(false, af, false, f1, (short)0, acc1, false, false);
    acc2 = __builtin_amdgcn_wmma_f32_16x16x32_bf16(false, af, false, f2, (short)0, acc2, false, false);
    acc3 = __builtin_amdgcn_wmma_f32_16x16x32_bf16(false, af, false, f3, (short)0, acc3, false, false);
  }
  const int nb = n0 + (lane & 15);
  const int mb = m0 + (sel << 3);
  v8f accs[4] = {acc0, acc1, acc2, acc3};
#pragma unroll
  for (int j = 0; j < 4; ++j) {
    int n = nb + j * 16;
    float b = bias ? bias[n] : 0.f;
#pragma unroll
    for (int r = 0; r < 8; ++r) {
      int m = mb + r;
      if (m < M) {
        float v = accs[j][r] + b;
        if (act == 1) v = lrelu_f(v);
        C[(size_t)m * Nout + n] = v;
        if (Cb) Cb[(size_t)m * Nout + n] = (__bf16)v;
      }
    }
  }
}

// ---------------- WMMA GEMM (fp32 A, bf16 W): stage 0 only ----------------
__global__ __launch_bounds__(256)
void gemm_wmma_fh(const float* __restrict__ A, int lda,
                  const __bf16* __restrict__ Wt, int ldw,
                  const float* __restrict__ bias,
                  float* __restrict__ C, __bf16* __restrict__ Cb,
                  int M, int Nout, int K, int act) {
  const int wave = threadIdx.x >> 5;
  const int lane = threadIdx.x & 31;
  const int m0 = (blockIdx.x * 8 + wave) * 16;
  const int n0 = blockIdx.y * 64;
  const int sel = lane >> 4;
  const int ra = min(m0 + (lane & 15), M - 1);
  const float* arow = A + (size_t)ra * lda;
  const __bf16* brow0 = Wt + (size_t)(n0 + (lane & 15)) * ldw;
  const __bf16* brow1 = brow0 + (size_t)16 * ldw;
  const __bf16* brow2 = brow0 + (size_t)32 * ldw;
  const __bf16* brow3 = brow0 + (size_t)48 * ldw;

  v8f acc0 = {}, acc1 = {}, acc2 = {}, acc3 = {};
  RawF rA  = load_raw_f(arow,  0, sel);
  RawH rB0 = load_raw_h(brow0, 0, sel);
  RawH rB1 = load_raw_h(brow1, 0, sel);
  RawH rB2 = load_raw_h(brow2, 0, sel);
  RawH rB3 = load_raw_h(brow3, 0, sel);
  for (int kk = 0; kk < K; kk += 32) {
    v16bf af = cvt_frag(rA);
    v16bf f0 = cat_frag(rB0);
    v16bf f1 = cat_frag(rB1);
    v16bf f2 = cat_frag(rB2);
    v16bf f3 = cat_frag(rB3);
    const int kn = (kk + 32 < K) ? kk + 32 : kk;
    rA  = load_raw_f(arow,  kn, sel);
    rB0 = load_raw_h(brow0, kn, sel);
    rB1 = load_raw_h(brow1, kn, sel);
    rB2 = load_raw_h(brow2, kn, sel);
    rB3 = load_raw_h(brow3, kn, sel);
    acc0 = __builtin_amdgcn_wmma_f32_16x16x32_bf16(false, af, false, f0, (short)0, acc0, false, false);
    acc1 = __builtin_amdgcn_wmma_f32_16x16x32_bf16(false, af, false, f1, (short)0, acc1, false, false);
    acc2 = __builtin_amdgcn_wmma_f32_16x16x32_bf16(false, af, false, f2, (short)0, acc2, false, false);
    acc3 = __builtin_amdgcn_wmma_f32_16x16x32_bf16(false, af, false, f3, (short)0, acc3, false, false);
  }
  const int nb = n0 + (lane & 15);
  const int mb = m0 + (sel << 3);
  v8f accs[4] = {acc0, acc1, acc2, acc3};
#pragma unroll
  for (int j = 0; j < 4; ++j) {
    int n = nb + j * 16;
    float b = bias ? bias[n] : 0.f;
#pragma unroll
    for (int r = 0; r < 8; ++r) {
      int m = mb + r;
      if (m < M) {
        float v = accs[j][r] + b;
        if (act == 1) v = lrelu_f(v);
        C[(size_t)m * Nout + n] = v;
        if (Cb) Cb[(size_t)m * Nout + n] = (__bf16)v;
      }
    }
  }
}

// ---------------- weight conversion / repack ----------------
__global__ void f32_to_bf16_k(const float* __restrict__ in, __bf16* __restrict__ out, size_t n) {
  for (size_t i = blockIdx.x * (size_t)blockDim.x + threadIdx.x; i < n;
       i += (size_t)gridDim.x * blockDim.x) out[i] = (__bf16)in[i];
}
// gate.lin1 (H x 257) -> bf16 (H x 256): drop edge-attr column, align rows
__global__ void repack_lin1_k(const float* __restrict__ lin1, __bf16* __restrict__ out) {
  int idx = blockIdx.x * blockDim.x + threadIdx.x;
  if (idx < Hdim * Hdim) out[idx] = (__bf16)lin1[(size_t)(idx >> 8) * 257 + (idx & 255)];
}

// ---------------- naive small GEMM (G-sized mats) ----------------
__global__ void naive_gemm_k(const float* __restrict__ A, int lda,
                             const float* __restrict__ Wt, int ldw,
                             const float* __restrict__ bias,
                             float* __restrict__ C, int M, int Nout, int K) {
  int idx = blockIdx.x * blockDim.x + threadIdx.x;
  if (idx >= M * Nout) return;
  int m = idx / Nout, n = idx % Nout;
  float s = bias ? bias[n] : 0.f;
  for (int k = 0; k < K; ++k) s += A[(size_t)m * lda + k] * Wt[(size_t)n * ldw + k];
  C[idx] = s;
}

// ---------------- per-row dot products (wave per row) ----------------
__global__ void rowdot2_k(const float* __restrict__ A, const float* __restrict__ v1,
                          const float* __restrict__ v2, float* __restrict__ o1,
                          float* __restrict__ o2, int M) {
  int row = blockIdx.x * 8 + (threadIdx.x >> 5);
  if (row >= M) return;
  int lane = threadIdx.x & 31;
  const float* a = A + (size_t)row * Hdim;
  float s1 = 0.f, s2 = 0.f;
#pragma unroll
  for (int j = 0; j < Hdim / 32; ++j) {
    float av = a[lane + 32 * j];
    s1 += av * v1[lane + 32 * j];
    if (v2) s2 += av * v2[lane + 32 * j];
  }
#pragma unroll
  for (int off = 16; off; off >>= 1) { s1 += __shfl_xor(s1, off); s2 += __shfl_xor(s2, off); }
  if (lane == 0) { o1[row] = s1; if (o2) o2[row] = s2; }
}

// ---------------- GATE edge logits (wave per edge) ----------------
__global__ void gate_edge_logits_k(const float* __restrict__ hx,
                                   const float* __restrict__ xr,
                                   const float* __restrict__ lin1,  // (H x 257)
                                   const float* __restrict__ attl,
                                   const int* __restrict__ src, const int* __restrict__ dst,
                                   const float* __restrict__ ea,
                                   float* __restrict__ elog, int E) {
  int e = blockIdx.x * 8 + (threadIdx.x >> 5);
  if (e >= E) return;
  int lane = threadIdx.x & 31;
  int s = src[e];
  float eav = ea[e];
  float part = 0.f;
#pragma unroll
  for (int j = 0; j < Hdim / 32; ++j) {
    int i = lane + 32 * j;
    float he = lrelu_f(hx[(size_t)s * Hdim + i] + eav * lin1[(size_t)i * 257 + 256]);
    part += he * attl[i];
  }
#pragma unroll
  for (int off = 16; off; off >>= 1) part += __shfl_xor(part, off);
  if (lane == 0) elog[e] = lrelu_f(part + xr[dst[e]]);
}

// ---------------- GAT edge logits ----------------
__global__ void gat_edge_logits_k(const float* __restrict__ as_, const float* __restrict__ ad_,
                                  const int* __restrict__ src, const int* __restrict__ dst,
                                  float* __restrict__ elog, int E) {
  for (int i = blockIdx.x * blockDim.x + threadIdx.x; i < E; i += gridDim.x * blockDim.x)
    elog[i] = lrelu_f(as_[src[i]] + ad_[dst[i]]);
}

// node logits for mol phase
__global__ void node_logits_k(const float* __restrict__ as_, const float* __restrict__ adg,
                              const int* __restrict__ batch, float* __restrict__ out, int N) {
  for (int i = blockIdx.x * blockDim.x + threadIdx.x; i < N; i += gridDim.x * blockDim.x)
    out[i] = lrelu_f(as_[i] + adg[batch[i]]);
}

// ---------------- segment softmax pieces ----------------
__global__ void seg_max_k(const float* __restrict__ v, const int* __restrict__ seg,
                          unsigned* __restrict__ mx, int n) {
  for (int i = blockIdx.x * blockDim.x + threadIdx.x; i < n; i += gridDim.x * blockDim.x)
    atomicMax(&mx[seg[i]], fenc(v[i]));
}
__global__ void seg_expsum_k(float* __restrict__ v, const int* __restrict__ seg,
                             const unsigned* __restrict__ mx, float* __restrict__ sm, int n) {
  for (int i = blockIdx.x * blockDim.x + threadIdx.x; i < n; i += gridDim.x * blockDim.x) {
    int s = seg[i];
    float ex = __expf(v[i] - fdec(mx[s]));
    v[i] = ex;
    atomicAdd(&sm[s], ex);
  }
}

// ---------------- edge message aggregation: block per edge ----------------
__global__ void edge_aggregate_k(const float* __restrict__ msg,
                                 const float* __restrict__ num,
                                 const float* __restrict__ den,
                                 const int* __restrict__ src, const int* __restrict__ dst,
                                 float* acc, int E) {
  int e = blockIdx.x;
  if (e >= E) return;
  int s = src[e], d = dst[e];
  float a = num[e] / (den[d] + 1e-16f);
  int t = threadIdx.x;
  atomicAdd(&acc[(size_t)d * Hdim + t], msg[(size_t)s * Hdim + t] * a);
}

// ---------------- graph pooling with LDS tile (8 x 256) ----------------
__global__ void pool_segment_k(const float* __restrict__ x, const int* __restrict__ batch,
                               const float* __restrict__ wnum,  // nullable per-node weight
                               const float* __restrict__ wden,  // nullable per-graph denom
                               float* acc, int N) {
  __shared__ float sh[Gnum * Hdim];
  int t = threadIdx.x;
#pragma unroll
  for (int g = 0; g < Gnum; ++g) sh[g * Hdim + t] = 0.f;
  __syncthreads();
  for (int n = blockIdx.x; n < N; n += gridDim.x) {
    int b = batch[n];
    float s = 1.f;
    if (wnum) s = wnum[n] / (wden[b] + 1e-16f);
    sh[b * Hdim + t] += x[(size_t)n * Hdim + t] * s;   // thread t owns column t: race-free
  }
  __syncthreads();
#pragma unroll
  for (int g = 0; g < Gnum; ++g) atomicAdd(&acc[g * Hdim + t], sh[g * Hdim + t]);
}

// ---------------- GRU elementwise (+ trailing relu); writes fp32 + bf16 mirror ----
__global__ void gru_elem_k(const float* __restrict__ gi, const float* __restrict__ gh,
                           const float* hprev, float* xout, __bf16* xmir, size_t total) {
  for (size_t idx = blockIdx.x * (size_t)blockDim.x + threadIdx.x; idx < total;
       idx += (size_t)gridDim.x * blockDim.x) {
    size_t m = idx >> 8;
    int i = (int)(idx & 255);
    const float* a = gi + m * 3 * Hdim;
    const float* b = gh + m * 3 * Hdim;
    float r  = sigm_f(a[i] + b[i]);
    float z  = sigm_f(a[Hdim + i] + b[Hdim + i]);
    float nn = tanhf(a[2 * Hdim + i] + r * b[2 * Hdim + i]);
    float v  = (1.f - z) * nn + z * hprev[idx];
    v = fmaxf(v, 0.f);
    xout[idx] = v;
    if (xmir) xmir[idx] = (__bf16)v;
  }
}

// ---------------- misc elementwise ----------------
__global__ void unary_k(float* p, __bf16* mir, size_t n, int mode) {  // 0=relu 1=elu
  for (size_t i = blockIdx.x * (size_t)blockDim.x + threadIdx.x; i < n;
       i += (size_t)gridDim.x * blockDim.x) {
    float v = p[i];
    v = (mode == 0) ? fmaxf(v, 0.f) : (v > 0.f ? v : __expf(v) - 1.f);
    p[i] = v;
    if (mir) mir[i] = (__bf16)v;
  }
}
__global__ void fill_f32_k(float* p, float v, size_t n) {
  for (size_t i = blockIdx.x * (size_t)blockDim.x + threadIdx.x; i < n;
       i += (size_t)gridDim.x * blockDim.x) p[i] = v;
}
__global__ void fill_u32_k(unsigned* p, unsigned v, size_t n) {
  for (size_t i = blockIdx.x * (size_t)blockDim.x + threadIdx.x; i < n;
       i += (size_t)gridDim.x * blockDim.x) p[i] = v;
}
__global__ void bcast_rows_k(float* p, const float* __restrict__ bias, int M) {
  size_t idx = blockIdx.x * (size_t)blockDim.x + threadIdx.x;
  if (idx < (size_t)M * Hdim) p[idx] = bias[idx & 255];
}
__global__ void gdot_k(const float* __restrict__ A, const float* __restrict__ v,
                       float* __restrict__ o, int M) {
  int m = blockIdx.x * blockDim.x + threadIdx.x;
  if (m >= M) return;
  float s = 0.f;
  for (int k = 0; k < Hdim; ++k) s += A[(size_t)m * Hdim + k] * v[k];
  o[m] = s;
}

// =========================================================================
extern "C" void kernel_launch(void* const* d_in, const int* in_sizes, int n_in,
                              void* d_out, int out_size, void* d_ws, size_t ws_size,
                              hipStream_t stream) {
  (void)n_in; (void)out_size; (void)ws_size;
  // -------- inputs (setup_inputs dict order, params flattened as jax pytree:
  //          nested dict keys sorted alphabetically, lists in order) --------
  const float* x_in  = (const float*)d_in[0];
  const int*   src   = (const int*)d_in[1];
  const int*   dst   = (const int*)d_in[2];
  const float* eattr = (const float*)d_in[3];
  const int*   batch = (const int*)d_in[4];
  const int N   = in_sizes[4];
  const int E   = in_sizes[1];
  const int FIN = in_sizes[0] / N;

  const float* P[64];
  for (int i = 5; i < 42; ++i) P[i] = (const float*)d_in[i];
  const float* ac_att_dst[2] = {P[5],  P[9]};
  const float* ac_att_src[2] = {P[6],  P[10]};
  const float* ac_bias[2]    = {P[7],  P[11]};
  const float* ac_lin[2]     = {P[8],  P[12]};
  const float* ag_wih[2] = {P[13], P[17]};
  const float* ag_whh[2] = {P[14], P[18]};
  const float* ag_bih[2] = {P[15], P[19]};
  const float* ag_bhh[2] = {P[16], P[20]};
  const float* g_att_l = P[21], *g_att_r = P[22], *g_bias = P[23];
  const float* g_lin1  = P[24], *g_lin2  = P[25];
  const float* gr0_wih = P[26], *gr0_whh = P[27], *gr0_bih = P[28], *gr0_bhh = P[29];
  const float* lin_in_b = P[30], *lin_in_w = P[31];
  const float* lin_out_b = P[32], *lin_out_w = P[33];
  const float* m_att_dst = P[34], *m_att_src = P[35], *m_bias = P[36], *m_lin = P[37];
  const float* mg_wih = P[38], *mg_whh = P[39], *mg_bih = P[40], *mg_bhh = P[41];

  // -------- workspace carve: fp32 masters, then bf16 mirrors/weights --------
  const size_t NH = (size_t)N * Hdim;
  const size_t HH = (size_t)Hdim * Hdim;
  float* Xb  = (float*)d_ws;        // N x H   node features (fp32 master)
  float* HBb = Xb  + NH;            // N x H   conv output h
  float* T0  = HBb + NH;            // N x 3H  scratch (hx / hs / gi)
  float* T1  = T0  + 3 * NH;        // N x 3H  scratch (lin2x / gh)
  float* EL  = T1  + 3 * NH;        // E       edge logits -> exp
  float* NA  = EL  + E;             // N
  float* NB  = NA  + N;             // N
  float* NS  = NB  + N;             // N       softmax denominators
  unsigned* NM = (unsigned*)(NS + N);            // N  encoded maxima
  float* OUTB = (float*)(NM + N);   // G x H
  float* HG   = OUTB + Gnum * Hdim; // G x H
  float* HD   = HG   + Gnum * Hdim; // G x H
  float* GI   = HD   + Gnum * Hdim; // G x 3H
  float* GH2  = GI   + Gnum * 3 * Hdim;
  float* ADG  = GH2  + Gnum * 3 * Hdim;          // G
  float* GS   = ADG  + Gnum;                     // G
  unsigned* GM = (unsigned*)(GS + Gnum);         // G
  uintptr_t hb = ((uintptr_t)(GM + Gnum) + 15) & ~(uintptr_t)15;
  __bf16* Xh  = (__bf16*)hb;        // N x H  bf16 mirror of Xb
  __bf16* HBh = Xh + NH;            // N x H  bf16 mirror of HBb
  __bf16* WIN = HBh + NH;           // H x FIN
  __bf16* WL1 = WIN + (size_t)Hdim * FIN;  // H x H (repacked gate.lin1)
  __bf16* WL2 = WL1 + HH;           // H x H
  __bf16* W0I = WL2 + HH;           // 3H x H
  __bf16* W0H = W0I + 3 * HH;
  __bf16* WA0 = W0H + 3 * HH;       // H x H
  __bf16* WA1 = WA0 + HH;
  __bf16* WI0 = WA1 + HH;           // 3H x H
  __bf16* WH0 = WI0 + 3 * HH;
  __bf16* WI1 = WH0 + 3 * HH;
  __bf16* WH1 = WI1 + 3 * HH;
  __bf16* WML = WH1 + 3 * HH;       // H x H

  const dim3 B256(256);
  auto gemm = [&](const __bf16* A, int lda, const __bf16* Wt, int ldw, const float* bias,
                  float* C, __bf16* Cb, int M, int Nout, int K, int act) {
    dim3 grid((M + 127) / 128, Nout / 64);
    gemm_wmma_hh<<<grid, B256, 0, stream>>>(A, lda, Wt, ldw, bias, C, Cb, M, Nout, K, act);
  };
  auto cvt = [&](const float* s, __bf16* d, size_t n) {
    f32_to_bf16_k<<<1024, B256, 0, stream>>>(s, d, n);
  };
  auto edge_softmax = [&]() {
    fill_u32_k<<<2048, B256, 0, stream>>>(NM, ENC_NEG_INF, (size_t)N);
    fill_f32_k<<<2048, B256, 0, stream>>>(NS, 0.f, (size_t)N);
    seg_max_k<<<2048, B256, 0, stream>>>(EL, dst, NM, E);
    seg_expsum_k<<<2048, B256, 0, stream>>>(EL, dst, NM, NS, E);
  };

  // ===== weights -> bf16 (once per launch; bit-identical to in-loop cvt) =====
  cvt(lin_in_w, WIN, (size_t)Hdim * FIN);
  repack_lin1_k<<<(int)((HH + 255) / 256), B256, 0, stream>>>(g_lin1, WL1);
  cvt(g_lin2, WL2, HH);
  cvt(gr0_wih, W0I, 3 * HH);  cvt(gr0_whh, W0H, 3 * HH);
  cvt(ac_lin[0], WA0, HH);    cvt(ac_lin[1], WA1, HH);
  cvt(ag_wih[0], WI0, 3 * HH); cvt(ag_whh[0], WH0, 3 * HH);
  cvt(ag_wih[1], WI1, 3 * HH); cvt(ag_whh[1], WH1, 3 * HH);
  cvt(m_lin, WML, HH);

  // ===== stage 0: x = lrelu(x_in @ Win^T + bin)  (fp32 A path) =====
  {
    dim3 grid((N + 127) / 128, Hdim / 64);
    gemm_wmma_fh<<<grid, B256, 0, stream>>>(x_in, FIN, WIN, FIN, lin_in_b,
                                            Xb, Xh, N, Hdim, FIN, /*lrelu*/1);
  }

  // ===== stage 1: GATEConv =====
  gemm(Xh, Hdim, WL1, Hdim, nullptr, T0, nullptr, N, Hdim, Hdim, 0);  // hx
  gemm(Xh, Hdim, WL2, Hdim, nullptr, T1, nullptr, N, Hdim, Hdim, 0);  // lin2(x)
  rowdot2_k<<<(N + 7) / 8, B256, 0, stream>>>(Xb, g_att_r, nullptr, NA, nullptr, N);
  gate_edge_logits_k<<<(E + 7) / 8, B256, 0, stream>>>(T0, NA, g_lin1, g_att_l,
                                                       src, dst, eattr, EL, E);
  edge_softmax();
  bcast_rows_k<<<(unsigned)((NH + 255) / 256), B256, 0, stream>>>(HBb, g_bias, N);
  edge_aggregate_k<<<E, B256, 0, stream>>>(T1, EL, NS, src, dst, HBb, E);
  unary_k<<<4096, B256, 0, stream>>>(HBb, HBh, NH, /*elu*/1);

  // ===== stage 2: x = relu(gru0(h, x)) =====
  gemm(HBh, Hdim, W0I, Hdim, gr0_bih, T0, nullptr, N, 3 * Hdim, Hdim, 0);
  gemm(Xh,  Hdim, W0H, Hdim, gr0_bhh, T1, nullptr, N, 3 * Hdim, Hdim, 0);
  gru_elem_k<<<4096, B256, 0, stream>>>(T0, T1, Xb, Xb, Xh, NH);

  // ===== stage 3: atom GAT layers =====
  const __bf16* WA[2] = {WA0, WA1};
  const __bf16* WI[2] = {WI0, WI1};
  const __bf16* WH[2] = {WH0, WH1};
  for (int l = 0; l < 2; ++l) {
    gemm(Xh, Hdim, WA[l], Hdim, nullptr, T0, nullptr, N, Hdim, Hdim, 0);  // h_src
    rowdot2_k<<<(N + 7) / 8, B256, 0, stream>>>(T0, ac_att_src[l], ac_att_dst[l], NA, NB, N);
    gat_edge_logits_k<<<2048, B256, 0, stream>>>(NA, NB, src, dst, EL, E);
    edge_softmax();
    bcast_rows_k<<<(unsigned)((NH + 255) / 256), B256, 0, stream>>>(HBb, ac_bias[l], N);
    edge_aggregate_k<<<E, B256, 0, stream>>>(T0, EL, NS, src, dst, HBb, E);
    unary_k<<<4096, B256, 0, stream>>>(HBb, HBh, NH, 1);
    gemm(HBh, Hdim, WI[l], Hdim, ag_bih[l], T0, nullptr, N, 3 * Hdim, Hdim, 0);
    gemm(Xh,  Hdim, WH[l], Hdim, ag_bhh[l], T1, nullptr, N, 3 * Hdim, Hdim, 0);
    gru_elem_k<<<4096, B256, 0, stream>>>(T0, T1, Xb, Xb, Xh, NH);
  }

  // ===== stage 4: out = relu(segment_sum(x, batch)) =====
  fill_f32_k<<<8, B256, 0, stream>>>(OUTB, 0.f, (size_t)Gnum * Hdim);
  pool_segment_k<<<256, B256, 0, stream>>>(Xb, batch, nullptr, nullptr, OUTB, N);
  unary_k<<<8, B256, 0, stream>>>(OUTB, nullptr, (size_t)Gnum * Hdim, 0);

  // ===== stage 5: mol readout (h_src, att_src loop-invariant) =====
  gemm(Xh, Hdim, WML, Hdim, nullptr, T0, nullptr, N, Hdim, Hdim, 0);   // h_src
  rowdot2_k<<<(N + 7) / 8, B256, 0, stream>>>(T0, m_att_src, nullptr, NA, nullptr, N);
  for (int t = 0; t < 2; ++t) {
    naive_gemm_k<<<(Gnum * Hdim + 255) / 256, B256, 0, stream>>>(
        OUTB, Hdim, m_lin, Hdim, nullptr, HD, Gnum, Hdim, Hdim);       // h_dst
    gdot_k<<<1, Gnum, 0, stream>>>(HD, m_att_dst, ADG, Gnum);
    node_logits_k<<<2048, B256, 0, stream>>>(NA, ADG, batch, EL, N);
    fill_u32_k<<<1, Gnum, 0, stream>>>(GM, ENC_NEG_INF, (size_t)Gnum);
    fill_f32_k<<<1, Gnum, 0, stream>>>(GS, 0.f, (size_t)Gnum);
    seg_max_k<<<2048, B256, 0, stream>>>(EL, batch, GM, N);
    seg_expsum_k<<<2048, B256, 0, stream>>>(EL, batch, GM, GS, N);
    bcast_rows_k<<<(Gnum * Hdim + 255) / 256, B256, 0, stream>>>(HG, m_bias, Gnum);
    pool_segment_k<<<256, B256, 0, stream>>>(T0, batch, EL, GS, HG, N);
    unary_k<<<8, B256, 0, stream>>>(HG, nullptr, (size_t)Gnum * Hdim, 1);  // elu
    naive_gemm_k<<<(Gnum * 3 * Hdim + 255) / 256, B256, 0, stream>>>(
        HG, Hdim, mg_wih, Hdim, mg_bih, GI, Gnum, 3 * Hdim, Hdim);
    naive_gemm_k<<<(Gnum * 3 * Hdim + 255) / 256, B256, 0, stream>>>(
        OUTB, Hdim, mg_whh, Hdim, mg_bhh, GH2, Gnum, 3 * Hdim, Hdim);
    gru_elem_k<<<8, B256, 0, stream>>>(GI, GH2, OUTB, OUTB, nullptr, (size_t)Gnum * Hdim);
  }

  // ===== stage 6: d_out = out @ lin_out^T + b =====
  naive_gemm_k<<<(Gnum * OUTdim + 255) / 256, B256, 0, stream>>>(
      OUTB, Hdim, lin_out_w, Hdim, lin_out_b, (float*)d_out, Gnum, OUTdim, Hdim);
}
